// GAT_1382979470033
// MI455X (gfx1250) — matrix operand order repaired and checked by reading, not compiled, not run on hardware
//
#include <hip/hip_runtime.h>
#include <math.h>

// ---------------------------------------------------------------------------
// 2-layer GAT for MI455X (gfx1250, wave32).
//  - GEMMs: v_wmma_f32_16x16x32_f16. One wave = 16-row panel x all N tiles.
//    A streamed as b128 loads (f32->f16 in-reg, prefetch-ahead); B pre-packed
//    to f16 fragment layout (zero K-padding) and staged ONCE per workgroup
//    into LDS -> per-step fragments come from ds_load_b128, vmem is A-only.
//  - Edge phase: L2-resident f32 atomics; fused softmax denominator +
//    message aggregation (segment_sum(p*xw)/segment_sum(p)).
// ---------------------------------------------------------------------------

typedef __attribute__((ext_vector_type(16))) _Float16 v16h;
typedef __attribute__((ext_vector_type(8)))  float    v8f;

static constexpr int NODES = 100000;
static constexpr int EDGES = 1600000;
static constexpr int FIN   = 500;
static constexpr int H1 = 8,  C1 = 8;    // layer 1: 8 heads x 8 dims, concat
static constexpr int H2 = 1,  C2 = 16;   // layer 2: 1 head x 16 classes, mean
static constexpr float NEG_SLOPE = 0.2f;

// monotonic float->uint map of -INFINITY: bits 0xFF800000 (negative) -> ~u
static constexpr unsigned NEGINF_MONO = 0x007FFFFFu;

__device__ __forceinline__ unsigned f2mono(float f) {
  unsigned u = __float_as_uint(f);
  return (u & 0x80000000u) ? ~u : (u | 0x80000000u);
}
__device__ __forceinline__ float mono2f(unsigned u) {
  return __uint_as_float((u & 0x80000000u) ? (u ^ 0x80000000u) : ~u);
}

// ---------------------------------------------------------------------------
// Pack B[K x NC] (f32 row-major) into per-K-step WMMA B fragments (f16),
// zero-padded for k >= KDIM. Layout: Bp[((s*NT + tn)*32 + lane)*16 + i],
// fragment element i at lane L: col = tn*16 + (L&15), k = s*32+(L>>4)*8+i+(i>=8?8:0).
// ---------------------------------------------------------------------------
template <int KDIM, int NC>
__global__ void pack_B(const float* __restrict__ B, _Float16* __restrict__ Bp) {
  constexpr int NT = NC / 16;
  constexpr int KSTEPS = (KDIM + 31) / 32;
  const int t = blockIdx.x * blockDim.x + threadIdx.x;  // (s*NT+tn)*32 + lane
  if (t >= KSTEPS * NT * 32) return;
  const int lane = t & 31;
  const int tile = t >> 5;
  const int tn = tile % NT;
  const int s  = tile / NT;
  const int col   = tn * 16 + (lane & 15);
  const int kbase = s * 32 + (lane >> 4) * 8;
  _Float16* dst = Bp + (size_t)t * 16;
#pragma unroll
  for (int i = 0; i < 16; ++i) {
    const int k = kbase + (i < 8 ? i : i + 8);
    dst[i] = (k < KDIM) ? (_Float16)B[(size_t)k * NC + col] : (_Float16)0.0f;
  }
}

// ---------------------------------------------------------------------------
// WMMA GEMM: C[M x NC] = A[M x K] * Bp; one wave computes rows [tm*16,tm*16+16)
// across ALL NC columns (NT tiles), reusing each A fragment NT times.
// Packed B panel staged once per workgroup in LDS (ds_load_b128 thereafter).
// ---------------------------------------------------------------------------
template <int KDIM, int NC>
__global__ void wmma_gemm_f16(const float* __restrict__ A,
                              const _Float16* __restrict__ Bp,
                              float* __restrict__ C, int M) {
  constexpr int NT     = NC / 16;
  constexpr int KFULL  = KDIM / 32;          // full (unpredicated) K steps
  constexpr bool TAIL  = (KDIM % 32) != 0;
  constexpr int KSTEPS = KFULL + (TAIL ? 1 : 0);
  constexpr int TOT_H  = KSTEPS * NT * 512;  // packed halves (64KB for layer 1)

  __shared__ __align__(32) _Float16 sB[TOT_H];
  {
    const int4* src = (const int4*)Bp;       // 16B = 8 halves per copy
    int4* dst = (int4*)sB;
    for (int i = threadIdx.x; i < TOT_H / 8; i += blockDim.x) dst[i] = src[i];
  }
  __syncthreads();

  const int tm = (int)((blockIdx.x * blockDim.x + threadIdx.x) >> 5);
  if (tm * 16 >= M) return;
  const int lane = threadIdx.x & 31;
  const int r  = lane & 15;   // A row / B,C col within tile
  const int hi = lane >> 4;   // half-wave selector

  v8f acc[NT];
#pragma unroll
  for (int t = 0; t < NT; ++t) acc[t] = (v8f){};

  const float* arow = A + (size_t)(tm * 16 + r) * KDIM;
  const _Float16* bl = sB + (size_t)lane * 16;  // + (s*NT+tn)*512

  for (int s = 0; s < KFULL; ++s) {
    const int k0 = s * 32;
    // A fragment: chunk0 = k0+hi*8 .. +7 ; chunk1 = k0+16+hi*8 .. +7
    const float4 p0 = *(const float4*)(arow + k0 + hi * 8);
    const float4 p1 = *(const float4*)(arow + k0 + hi * 8 + 4);
    const float4 q0 = *(const float4*)(arow + k0 + 16 + hi * 8);
    const float4 q1 = *(const float4*)(arow + k0 + 16 + hi * 8 + 4);
    if (k0 + 64 < KDIM)  // stream-ahead hint -> global_prefetch_b8
      __builtin_prefetch(arow + k0 + 64, 0, 1);
    v16h af;
    af[0] = (_Float16)p0.x; af[1] = (_Float16)p0.y;
    af[2] = (_Float16)p0.z; af[3] = (_Float16)p0.w;
    af[4] = (_Float16)p1.x; af[5] = (_Float16)p1.y;
    af[6] = (_Float16)p1.z; af[7] = (_Float16)p1.w;
    af[8]  = (_Float16)q0.x; af[9]  = (_Float16)q0.y;
    af[10] = (_Float16)q0.z; af[11] = (_Float16)q0.w;
    af[12] = (_Float16)q1.x; af[13] = (_Float16)q1.y;
    af[14] = (_Float16)q1.z; af[15] = (_Float16)q1.w;
#pragma unroll
    for (int t = 0; t < NT; ++t) {
      const v16h bf = *(const v16h*)(bl + (size_t)(s * NT + t) * 512);
      acc[t] = __builtin_amdgcn_wmma_f32_16x16x32_f16(
          false, af, false, bf, (short)0, acc[t], false, false);
    }
  }

  if constexpr (TAIL) {
    // Packed B is zero for k >= KDIM, so A values there are don't-cares:
    // clamp the address (no exec-mask predication), value multiplied by 0.
    const int k0 = KFULL * 32;
    v16h af;
#pragma unroll
    for (int i = 0; i < 16; ++i) {
      const int vg = i >> 1, sl = i & 1;
      const int k = (vg < 4) ? (vg * 2 + sl + hi * 8)
                             : ((vg - 4) * 2 + sl + 16 + hi * 8);
      const int kk = k0 + k;
      af[i] = (_Float16)arow[kk < KDIM ? kk : 0];
    }
#pragma unroll
    for (int t = 0; t < NT; ++t) {
      const v16h bf = *(const v16h*)(bl + (size_t)(KFULL * NT + t) * 512);
      acc[t] = __builtin_amdgcn_wmma_f32_16x16x32_f16(
          false, af, false, bf, (short)0, acc[t], false, false);
    }
  }

  // C fragment: VGPR v -> M = v + 8*hi, N = r (per 16x16 tile tn)
#pragma unroll
  for (int t = 0; t < NT; ++t) {
#pragma unroll
    for (int v = 0; v < 8; ++v) {
      const int m = tm * 16 + v + hi * 8;
      C[(size_t)m * NC + t * 16 + r] = acc[t][v];
    }
  }
}

// ---- per-(node,head) attention half-scores: s = <xw[n,h,:], a[h,:]> --------
template <int HEADS, int CD>
__global__ void node_scores(const float* __restrict__ xw,
                            const float* __restrict__ a_src,
                            const float* __restrict__ a_dst,
                            float* __restrict__ s_src,
                            float* __restrict__ s_dst) {
  const int idx = blockIdx.x * blockDim.x + threadIdx.x;  // node*HEADS + h
  if (idx >= NODES * HEADS) return;
  const int h = idx % HEADS;
  const float* v = xw + (size_t)(idx / HEADS) * HEADS * CD + h * CD;
  float ss = 0.0f, sd = 0.0f;
#pragma unroll
  for (int c = 0; c < CD; ++c) {
    ss += v[c] * a_src[h * CD + c];
    sd += v[c] * a_dst[h * CD + c];
  }
  s_src[idx] = ss;
  s_dst[idx] = sd;
}

// ---- buffer init ----------------------------------------------------------
__global__ void fill_f32(float* p, float v, long n) {
  const long i = (long)blockIdx.x * blockDim.x + threadIdx.x;
  if (i < n) p[i] = v;
}
__global__ void fill_u32(unsigned* p, unsigned v, long n) {
  const long i = (long)blockIdx.x * blockDim.x + threadIdx.x;
  if (i < n) p[i] = v;
}

// ---- pass 1 over edges: segment-max of leaky_relu(s_src[s]+s_dst[d]) ------
template <int HEADS>
__global__ void edge_max(const int* __restrict__ ei,
                         const float* __restrict__ s_src,
                         const float* __restrict__ s_dst,
                         unsigned* __restrict__ maxb) {
  const long t = (long)blockIdx.x * blockDim.x + threadIdx.x;
  const long total = (long)(EDGES + NODES) * HEADS;
  if (t >= total) return;
  const int e = (int)(t / HEADS), h = (int)(t - (long)e * HEADS);
  int s, d;
  if (e < EDGES) { s = ei[e]; d = ei[EDGES + e]; } else { s = d = e - EDGES; }
  float v = s_src[(size_t)s * HEADS + h] + s_dst[(size_t)d * HEADS + h];
  v = (v > 0.0f) ? v : NEG_SLOPE * v;
  atomicMax(maxb + (size_t)d * HEADS + h, f2mono(v));
}

// ---- pass 2 over edges: p = exp(e-m); denom[d] += p; acc[d] += p*xw[s] ----
template <int HEADS, int CD>
__global__ void edge_acc(const int* __restrict__ ei,
                         const float* __restrict__ s_src,
                         const float* __restrict__ s_dst,
                         const unsigned* __restrict__ maxb,
                         const float* __restrict__ xw,
                         float* __restrict__ denom,
                         float* __restrict__ acc) {
  const long t = (long)blockIdx.x * blockDim.x + threadIdx.x;
  const long total = (long)(EDGES + NODES) * HEADS;
  if (t >= total) return;
  const int e = (int)(t / HEADS), h = (int)(t - (long)e * HEADS);
  int s, d;
  if (e < EDGES) { s = ei[e]; d = ei[EDGES + e]; } else { s = d = e - EDGES; }
  float v = s_src[(size_t)s * HEADS + h] + s_dst[(size_t)d * HEADS + h];
  v = (v > 0.0f) ? v : NEG_SLOPE * v;
  const float m = mono2f(maxb[(size_t)d * HEADS + h]);
  const float p = expf(v - m);
  atomicAdd(denom + (size_t)d * HEADS + h, p);
  const float* xs = xw + ((size_t)s * HEADS + h) * CD;
  float* ad = acc + ((size_t)d * HEADS + h) * CD;
#pragma unroll
  for (int c = 0; c < CD; ++c) atomicAdd(ad + c, p * xs[c]);
}

// ---- layer-1 finish: h = elu(acc/denom + b1), concat layout ---------------
__global__ void finish1(const float* __restrict__ acc,
                        const float* __restrict__ denom,
                        const float* __restrict__ b1,
                        float* __restrict__ hout) {
  const long idx = (long)blockIdx.x * blockDim.x + threadIdx.x;
  if (idx >= (long)NODES * H1 * C1) return;
  const int j = (int)(idx % (H1 * C1));
  const long node = idx / (H1 * C1);
  const float v = acc[idx] / denom[node * H1 + j / C1] + b1[j];
  hout[idx] = (v > 0.0f) ? v : (expf(v) - 1.0f);   // ELU
}

// ---- layer-2 finish: out = log_softmax(acc/denom + b2) --------------------
__global__ void finish2(const float* __restrict__ acc,
                        const float* __restrict__ denom,
                        const float* __restrict__ b2,
                        float* __restrict__ out) {
  const int node = blockIdx.x * blockDim.x + threadIdx.x;
  if (node >= NODES) return;
  const float inv = 1.0f / denom[node];
  float v[C2], mx = -INFINITY;
#pragma unroll
  for (int c = 0; c < C2; ++c) {
    v[c] = acc[(size_t)node * C2 + c] * inv + b2[c];
    mx = fmaxf(mx, v[c]);
  }
  float se = 0.0f;
#pragma unroll
  for (int c = 0; c < C2; ++c) se += expf(v[c] - mx);
  const float lse = mx + logf(se);
#pragma unroll
  for (int c = 0; c < C2; ++c) out[(size_t)node * C2 + c] = v[c] - lse;
}

// ---------------------------------------------------------------------------
static inline int nblk(long n, int b) { return (int)((n + b - 1) / b); }

extern "C" void kernel_launch(void* const* d_in, const int* in_sizes, int n_in,
                              void* d_out, int out_size, void* d_ws, size_t ws_size,
                              hipStream_t stream) {
  const float* x      = (const float*)d_in[0];
  const int*   ei     = (const int*)  d_in[1];   // [2,E] int32 (jax x64 off)
  const float* W1     = (const float*)d_in[2];
  const float* a_src1 = (const float*)d_in[3];
  const float* a_dst1 = (const float*)d_in[4];
  const float* b1     = (const float*)d_in[5];
  const float* W2     = (const float*)d_in[6];
  const float* a_src2 = (const float*)d_in[7];
  const float* a_dst2 = (const float*)d_in[8];
  const float* b2     = (const float*)d_in[9];
  float* out = (float*)d_out;

  // ---- workspace bump layout (floats), peak ~64 MB + 66 KB ----
  float* ws = (float*)d_ws;
  float*    xw    = ws;                         // [N,64] xw1; later reused as h
  float*    ssrc  = xw   + (size_t)NODES * 64;  // [N,8]
  float*    sdst  = ssrc + (size_t)NODES * 8;   // [N,8]
  unsigned* maxb  = (unsigned*)(sdst + (size_t)NODES * 8);  // [N,8]
  float*    denom = (float*)maxb + (size_t)NODES * 8;       // [N,8]
  float*    acc   = denom + (size_t)NODES * 8;  // [N,64] acc1; layer2 overlays
  float*    xw2   = acc;                        // [N,16] (acc1 dead after finish1)
  float*    acc2  = acc + (size_t)NODES * C2;   // [N,16]
  _Float16* bp1   = (_Float16*)(acc + (size_t)NODES * 64);  // 16*4*32*16 f16
  _Float16* bp2   = bp1 + 16 * 4 * 32 * 16;                 //  2*1*32*16 f16

  const long etot1 = (long)(EDGES + NODES) * H1;
  const long etot2 = (long)(EDGES + NODES) * H2;

  // ---- pack weights into WMMA B-fragment layout (f16, zero K-padding) ----
  pack_B<FIN, 64><<<nblk(16L * 4 * 32, 256), 256, 0, stream>>>(W1, bp1);
  pack_B<64, 16><<<1, 64, 0, stream>>>(W2, bp2);

  // ================= layer 1 =================
  wmma_gemm_f16<FIN, 64><<<nblk(6250L * 32, 256), 256, 0, stream>>>(
      x, bp1, xw, NODES);
  node_scores<H1, C1><<<nblk((long)NODES * H1, 256), 256, 0, stream>>>(
      xw, a_src1, a_dst1, ssrc, sdst);
  fill_u32<<<nblk((long)NODES * H1, 256), 256, 0, stream>>>(
      maxb, NEGINF_MONO, (long)NODES * H1);
  fill_f32<<<nblk((long)NODES * H1, 256), 256, 0, stream>>>(
      denom, 0.0f, (long)NODES * H1);
  fill_f32<<<nblk((long)NODES * 64, 256), 256, 0, stream>>>(
      acc, 0.0f, (long)NODES * 64);
  edge_max<H1><<<nblk(etot1, 256), 256, 0, stream>>>(ei, ssrc, sdst, maxb);
  edge_acc<H1, C1><<<nblk(etot1, 256), 256, 0, stream>>>(
      ei, ssrc, sdst, maxb, xw, denom, acc);
  finish1<<<nblk((long)NODES * 64, 256), 256, 0, stream>>>(acc, denom, b1, xw);
  // xw now holds h = elu(gat1(x))

  // ================= layer 2 =================
  wmma_gemm_f16<64, 16><<<nblk(6250L * 32, 256), 256, 0, stream>>>(
      xw, bp2, xw2, NODES);
  node_scores<H2, C2><<<nblk((long)NODES * H2, 256), 256, 0, stream>>>(
      xw2, a_src2, a_dst2, ssrc, sdst);
  fill_u32<<<nblk((long)NODES, 256), 256, 0, stream>>>(
      maxb, NEGINF_MONO, (long)NODES);
  fill_f32<<<nblk((long)NODES, 256), 256, 0, stream>>>(
      denom, 0.0f, (long)NODES);
  fill_f32<<<nblk((long)NODES * C2, 256), 256, 0, stream>>>(
      acc2, 0.0f, (long)NODES * C2);
  edge_max<H2><<<nblk(etot2, 256), 256, 0, stream>>>(ei, ssrc, sdst, maxb);
  edge_acc<H2, C2><<<nblk(etot2, 256), 256, 0, stream>>>(
      ei, ssrc, sdst, maxb, xw2, denom, acc2);
  finish2<<<nblk((long)NODES, 256), 256, 0, stream>>>(acc2, denom, b2, out);
}
